// CoordfnTopologyLayer_3006477107665
// MI455X (gfx1250) — matrix operand order repaired and checked by reading, not compiled, not run on hardware
//
#include <hip/hip_runtime.h>
#include <hip/hip_bf16.h>

typedef __attribute__((ext_vector_type(16))) _Float16 v16h;
typedef __attribute__((ext_vector_type(8)))  float    v8f;
typedef __attribute__((ext_vector_type(4)))  unsigned int uint4v;
typedef __attribute__((ext_vector_type(4)))  int      int4v;
typedef __attribute__((ext_vector_type(8)))  int      int8v;

#define NN     100000
#define FIN    128
#define FHID   64
#define FF     8
#define FOUTN  128
#define EE     1600000
#define BB     200
#define NTILES 6250
#define WPB    4          // waves per block

// per-wave LDS floats: x tile (16 rows x 132 padded) + H (16 x 65) + fv (16 x 9)
#define XT_ROW   132
#define LDS_XT   (16 * XT_ROW)
#define LDS_H    (16 * 65)
#define LDS_FV   (16 * 9)
#define LDS_WAVE (LDS_XT + LDS_H + LDS_FV)

// workspace byte offsets (all 32B-aligned)
#define OFF_FV    0
#define OFF_WF1H  3200000
#define OFF_WF1L  (OFF_WF1H + 16384)
#define OFF_WF2H  (OFF_WF1L + 16384)
#define OFF_WF2L  (OFF_WF2H + 2048)
#define OFF_WOXH  (OFF_WF2L + 2048)
#define OFF_WOXL  (OFF_WOXH + 32768)
#define OFF_WOAH  (OFF_WOXL + 32768)
#define OFF_WOAL  (OFF_WOAH + 24576)

static __device__ __forceinline__ void split1(float v, _Float16 &h, _Float16 &l) {
  h = (_Float16)v;
  l = (_Float16)(v - (float)h);
}

static __device__ __forceinline__ v16h ld16h(const _Float16* p) {
  return *(const v16h*)p;
}

// D += A*B with A = ah+al, B = bh+bl (two-term f16 split, f32 accumulate)
static __device__ __forceinline__ v8f wmma3(v16h ah, v16h al, v16h bh, v16h bl, v8f c) {
  c = __builtin_amdgcn_wmma_f32_16x16x32_f16(false, al, false, bh, (short)0, c, false, false);
  c = __builtin_amdgcn_wmma_f32_16x16x32_f16(false, ah, false, bl, (short)0, c, false, false);
  c = __builtin_amdgcn_wmma_f32_16x16x32_f16(false, ah, false, bh, (short)0, c, false, false);
  return c;
}

// TDM: load a 16x128 f32 tile (row stride 128 elements) from global into LDS
// with 4-DWORD padding after each 128 DWORDs (LDS row stride = 132 floats).
static __device__ __forceinline__ void tdm_load_x_tile(const float* gsrc, unsigned lds_off) {
  const unsigned long long ga = (unsigned long long)(const void*)gsrc;

  uint4v g0;
  g0[0] = 1u;                                        // count=1, is_restore=0, gather=0
  g0[1] = lds_off;                                   // lds_addr [63:32]
  g0[2] = (unsigned)ga;                              // global_addr [95:64]
  g0[3] = ((unsigned)(ga >> 32) & 0x01FFFFFFu)       // global_addr [120:96]
        | 0x80000000u;                               // type=2 ("image") [127:126]

  int8v g1;
  g1[0] = (2 << 16)        // data_size = 2 -> 4 bytes
        | (1 << 20)        // pad_enable
        | (6 << 22)        // pad_interval code 6 -> every 128 DWORDs
        | (3 << 25);       // pad_amount  code 3 -> 4 DWORDs
  g1[1] = (FIN & 0xFFFF) << 16;            // tensor_dim0[15:0]  (=128)
  g1[2] = (FIN >> 16) | (16 << 16);        // tensor_dim0[31:16] | tensor_dim1[15:0] (=16)
  g1[3] = (16 >> 16) | (FIN << 16);        // tensor_dim1[31:16] | tile_dim0 (=128)
  g1[4] = 16;                              // tile_dim1 = 16, tile_dim2 = 0
  g1[5] = FIN;                             // tensor_dim0_stride[31:0] = 128
  g1[6] = 0;                               // stride0[47:32] | stride1[15:0]
  g1[7] = 0;                               // stride1[47:16]

  int4v g2 = {0, 0, 0, 0};
  int4v g3 = {0, 0, 0, 0};

#if defined(__clang_major__) && (__clang_major__ >= 23)
  int8v g4 = {0, 0, 0, 0, 0, 0, 0, 0};
  __builtin_amdgcn_tensor_load_to_lds(g0, g1, g2, g3, g4, 0);
#else
  __builtin_amdgcn_tensor_load_to_lds(g0, g1, g2, g3, 0);
#endif
}

// ---------------------------------------------------------------------------
// PREP: split f32 weights into hi/lo f16, transposed to B-fragment-friendly
// layout Wt[n][k] (contiguous in k). Wf2 padded N 8->16, Wout act-part padded
// K 72->96, all pads zero.
// ---------------------------------------------------------------------------
__global__ void prep_kernel(const float* __restrict__ Wf1,
                            const float* __restrict__ Wf2,
                            const float* __restrict__ Wout,
                            _Float16* __restrict__ wf1h, _Float16* __restrict__ wf1l,
                            _Float16* __restrict__ wf2h, _Float16* __restrict__ wf2l,
                            _Float16* __restrict__ woxh, _Float16* __restrict__ woxl,
                            _Float16* __restrict__ woah, _Float16* __restrict__ woal) {
  int idx = blockIdx.x * blockDim.x + threadIdx.x;
  if (idx < 8192) {                       // Wf1t: 64 x 128
    int n = idx >> 7, k = idx & 127;
    split1(Wf1[k * FHID + n], wf1h[idx], wf1l[idx]);
  } else if (idx < 9216) {                // Wf2t: 16 x 64 (n>=8 zero)
    int i = idx - 8192; int n = i >> 6, k = i & 63;
    float w = (n < FF) ? Wf2[k * FF + n] : 0.f;
    split1(w, wf2h[i], wf2l[i]);
  } else if (idx < 25600) {               // WoutXt: 128 x 128 (rows 0..127 of Wout)
    int i = idx - 9216; int n = i >> 7, k = i & 127;
    split1(Wout[k * FOUTN + n], woxh[i], woxl[i]);
  } else if (idx < 37888) {               // WoutAt: 128 x 96 (rows 128..199, k>=72 zero)
    int i = idx - 25600; int n = i / 96, k = i - n * 96;
    float w = (k < 72) ? Wout[(FIN + k) * FOUTN + n] : 0.f;
    split1(w, woah[i], woal[i]);
  }
}

// ---------------------------------------------------------------------------
// MAIN: one wave per 16-node tile. TDM stages the x tile into LDS, then fused:
//   H = relu(x@Wf1+bf1); fv = H@Wf2+bf2; act0 = coord_acts(fv);
//   out = relu([x,act0]@Wout + bout)
// All matmuls via v_wmma_f32_16x16x32_f16 with hi/lo split.
// ---------------------------------------------------------------------------
__global__ void __launch_bounds__(128)
main_kernel(const float* __restrict__ x,
            const float* __restrict__ bf1, const float* __restrict__ bf2,
            const float* __restrict__ tri_t, const float* __restrict__ gauss_c,
            const float* __restrict__ line_W, const float* __restrict__ line_b,
            const float* __restrict__ bout,
            float* __restrict__ fv,
            const _Float16* __restrict__ wf1h, const _Float16* __restrict__ wf1l,
            const _Float16* __restrict__ wf2h, const _Float16* __restrict__ wf2l,
            const _Float16* __restrict__ woxh, const _Float16* __restrict__ woxl,
            const _Float16* __restrict__ woah, const _Float16* __restrict__ woal,
            float* __restrict__ out) {
  __shared__ float lds[WPB * LDS_WAVE];

  const int lane = threadIdx.x & 31;
  const int wv   = threadIdx.x >> 5;
  const int tile = blockIdx.x * WPB + wv;
  if (tile >= NTILES) return;            // uniform per wave: EXEC all-1 inside

  float* xlds  = lds + wv * LDS_WAVE;
  float* Hlds  = xlds + LDS_XT;
  float* fvlds = Hlds + LDS_H;

  const int mrow = lane & 15;            // A-frag row / B-frag column
  const int hi   = (lane >> 4) & 1;
  const int base = hi * 8;               // C-row M offset of accumulator element r
  const int koff = hi * 16;              // B-frag K offset
  const int row0 = tile * 16;

  // ---- TDM: DMA the 16x128 f32 x-tile into LDS (padded rows of 132) ----
  tdm_load_x_tile(x + (size_t)row0 * FIN,
                  (unsigned)(size_t)(const void*)xlds);
  __builtin_amdgcn_s_wait_tensorcnt(0);
  asm volatile("" ::: "memory");

  // ---- x A-fragments from LDS (persist; reused by GEMM1 and GEMM3) ----
  const float* xr = xlds + mrow * XT_ROW;
  v16h xah[4], xal[4];
#pragma unroll
  for (int kc = 0; kc < 4; ++kc) {
    const float* p = xr + kc * 32 + koff;
#pragma unroll
    for (int h = 0; h < 8; ++h) {
      _Float16 a, b; split1(p[h], a, b);       xah[kc][h] = a;     xal[kc][h] = b;
    }
#pragma unroll
    for (int h = 0; h < 8; ++h) {
      _Float16 a, b; split1(p[8 + h], a, b);  xah[kc][8 + h] = a; xal[kc][8 + h] = b;
    }
  }

  // ---- GEMM1: acc1 = x @ Wf1  (16x128 * 128x64) ----
  v8f acc1[4] = {};
#pragma unroll
  for (int kc = 0; kc < 4; ++kc)
#pragma unroll
    for (int nc = 0; nc < 4; ++nc) {
      const int o = (nc * 16 + mrow) * 128 + kc * 32 + koff;
      acc1[nc] = wmma3(xah[kc], xal[kc], ld16h(wf1h + o), ld16h(wf1l + o), acc1[nc]);
    }

  // H = relu(acc1 + bf1) -> LDS (C-layout -> row-major for A-frag reload)
#pragma unroll
  for (int nc = 0; nc < 4; ++nc) {
    const int n = nc * 16 + mrow;
    const float bias = bf1[n];
#pragma unroll
    for (int r = 0; r < 8; ++r)
      Hlds[(r + base) * 65 + n] = fmaxf(acc1[nc][r] + bias, 0.f);
  }

  // ---- GEMM2: acc2 = H @ Wf2  (16x64 * 64x16, N padded) ----
  v8f acc2 = {};
#pragma unroll
  for (int kc = 0; kc < 2; ++kc) {
    v16h hh, hl;
    const float* hp = Hlds + mrow * 65 + kc * 32 + koff;
#pragma unroll
    for (int h = 0; h < 8; ++h) { _Float16 a, b; split1(hp[h], a, b);      hh[h] = a;     hl[h] = b; }
#pragma unroll
    for (int h = 0; h < 8; ++h) { _Float16 a, b; split1(hp[8 + h], a, b); hh[8 + h] = a; hl[8 + h] = b; }
    const int o = mrow * 64 + kc * 32 + koff;
    acc2 = wmma3(hh, hl, ld16h(wf2h + o), ld16h(wf2l + o), acc2);
  }

  // fv = acc2 + bf2 : store to global ws + LDS (for per-row act0)
  if (mrow < FF) {
    const float bias = bf2[mrow];
#pragma unroll
    for (int r = 0; r < 8; ++r) {
      const float v = acc2[r] + bias;
      fv[(size_t)(row0 + r + base) * FF + mrow] = v;
      fvlds[(r + base) * 9 + mrow] = v;
    }
  }

  float fvr[8];
#pragma unroll
  for (int f = 0; f < 8; ++f) fvr[f] = fvlds[mrow * 9 + f];

  // act0 parameters
  float t3[3], cg6[6], lsum[3], lb3[3];
#pragma unroll
  for (int j = 0; j < 3; ++j) {
    t3[j]   = tri_t[j];
    lsum[j] = line_W[j] + line_W[3 + j];   // birth==death -> W[0,j]+W[1,j]
    lb3[j]  = line_b[j];
  }
#pragma unroll
  for (int i = 0; i < 6; ++i) cg6[i] = gauss_c[i];

  // act0[node, f*9+j], p=(v,v): tri=relu(v-|t-v|), gau=exp(-((v-c0)^2+(v-c1)^2)*50), lin=v*(W0+W1)+b
  auto act0v = [&](int p) -> float {
    if (p >= 72) return 0.f;
    const int f = p / 9, j = p - f * 9;
    const float v = fvr[f];
    if (j < 3) return fmaxf(v - fabsf(t3[j] - v), 0.f);
    if (j < 6) {
      const int jj = j - 3;
      const float d0 = v - cg6[2 * jj], d1 = v - cg6[2 * jj + 1];
      return expf(-(d0 * d0 + d1 * d1) * 50.f);
    }
    const int jj = j - 6;
    return v * lsum[jj] + lb3[jj];
  };

  // ---- GEMM3: out = relu([x(K=128), act0(K=72 pad 96)] @ Wout + bout), N in 2 halves ----
#pragma unroll 1
  for (int nh = 0; nh < 2; ++nh) {
    v8f acc3[4] = {};
#pragma unroll
    for (int kc = 0; kc < 4; ++kc)
#pragma unroll
      for (int nc = 0; nc < 4; ++nc) {
        const int o = ((nh * 4 + nc) * 16 + mrow) * 128 + kc * 32 + koff;
        acc3[nc] = wmma3(xah[kc], xal[kc], ld16h(woxh + o), ld16h(woxl + o), acc3[nc]);
      }
#pragma unroll
    for (int c = 0; c < 3; ++c) {
      v16h aah, aal;
#pragma unroll
      for (int h = 0; h < 16; ++h) {
        const int p = c * 32 + koff + h;
        _Float16 a, b; split1(act0v(p), a, b);
        aah[h] = a; aal[h] = b;
      }
#pragma unroll
      for (int nc = 0; nc < 4; ++nc) {
        const int o = ((nh * 4 + nc) * 16 + mrow) * 96 + c * 32 + koff;
        acc3[nc] = wmma3(aah, aal, ld16h(woah + o), ld16h(woal + o), acc3[nc]);
      }
    }
#pragma unroll
    for (int nc = 0; nc < 4; ++nc) {
      const int n = (nh * 4 + nc) * 16 + mrow;
      const float bias = bout[n];
#pragma unroll
      for (int r = 0; r < 8; ++r)
        out[(size_t)(row0 + r + base) * FOUTN + n] = fmaxf(acc3[nc][r] + bias, 0.f);
    }
  }
}

// ---------------------------------------------------------------------------
// G1: closed form. (edge,f) pairs are unique, so
//   g1[b, f*9+j] = act1(0, v_f)_j + (M_b - 1) * az_j
// where az = act1(0,0) and M_b = #distinct masked edges in batch b.
// One wave per batch.
// ---------------------------------------------------------------------------
__global__ void g1_kernel(const float* __restrict__ fv,
                          const int* __restrict__ edge_index,
                          const int* __restrict__ random_edges,
                          const float* __restrict__ tri_t1,
                          const float* __restrict__ gauss_c1,
                          const float* __restrict__ line_W1,
                          const float* __restrict__ line_b1,
                          float* __restrict__ g1out) {
  __shared__ int   sh_e[8];
  __shared__ float sh_v[8];
  __shared__ int   sh_c[8];
  const int b = blockIdx.x;
  const int f = threadIdx.x;
  if (f < FF) {
    const int e = random_edges[b * FF + f];
    const int s = edge_index[e];
    const int d = edge_index[EE + e];
    sh_e[f] = e;
    sh_v[f] = fmaxf(fv[(size_t)s * FF + f], fv[(size_t)d * FF + f]);
  }
  __syncthreads();
  if (f < FF) {
    const int e = sh_e[f];
    bool masked = false, first = true;
    for (int f2 = 0; f2 < FF; ++f2)
      if (sh_e[f2] == e) {
        if (sh_v[f2] != 0.f) masked = true;
        if (f2 < f) first = false;
      }
    sh_c[f] = (first && masked) ? 1 : 0;
  }
  __syncthreads();
  if (f >= FF) return;
  int Mb = 0;
  for (int f2 = 0; f2 < FF; ++f2) Mb += sh_c[f2];
  const float v = sh_v[f];
  const float scale = (float)(Mb - 1);
  for (int j = 0; j < 3; ++j) {
    const float t  = tri_t1[j];
    const float c0 = gauss_c1[2 * j], c1 = gauss_c1[2 * j + 1];
    const float w1 = line_W1[3 + j], bb = line_b1[j];
    const float trv = fmaxf(v - fabsf(t), 0.f);            // act1(0,v) tri
    const float trz = fmaxf(0.f - fabsf(t), 0.f);          // az tri
    const float dv  = v - c1;
    const float gv  = expf(-(c0 * c0 + dv * dv) * 50.f);   // act1(0,v) gauss
    const float gz  = expf(-(c0 * c0 + c1 * c1) * 50.f);   // az gauss
    const float lv  = v * w1 + bb;                         // act1(0,v) lin
    const float lz  = bb;                                  // az lin
    g1out[b * 72 + f * 9 + j]     = trv + scale * trz;
    g1out[b * 72 + f * 9 + 3 + j] = gv  + scale * gz;
    g1out[b * 72 + f * 9 + 6 + j] = lv  + scale * lz;
  }
}

// ---------------------------------------------------------------------------
extern "C" void kernel_launch(void* const* d_in, const int* in_sizes, int n_in,
                              void* d_out, int out_size, void* d_ws, size_t ws_size,
                              hipStream_t stream) {
  const float* x       = (const float*)d_in[0];
  const float* Wf1     = (const float*)d_in[1];
  const float* bf1     = (const float*)d_in[2];
  const float* Wf2     = (const float*)d_in[3];
  const float* bf2     = (const float*)d_in[4];
  const float* tri_t   = (const float*)d_in[5];
  const float* gauss_c = (const float*)d_in[6];
  const float* line_W  = (const float*)d_in[7];
  const float* line_b  = (const float*)d_in[8];
  const float* tri_t1  = (const float*)d_in[9];
  const float* gauss_c1= (const float*)d_in[10];
  const float* line_W1 = (const float*)d_in[11];
  const float* line_b1 = (const float*)d_in[12];
  const float* Wout    = (const float*)d_in[13];
  const float* bout    = (const float*)d_in[14];
  const int*   edge_index   = (const int*)d_in[15];
  const int*   random_edges = (const int*)d_in[17];

  char* ws = (char*)d_ws;
  float*    fv   = (float*)(ws + OFF_FV);
  _Float16* wf1h = (_Float16*)(ws + OFF_WF1H);
  _Float16* wf1l = (_Float16*)(ws + OFF_WF1L);
  _Float16* wf2h = (_Float16*)(ws + OFF_WF2H);
  _Float16* wf2l = (_Float16*)(ws + OFF_WF2L);
  _Float16* woxh = (_Float16*)(ws + OFF_WOXH);
  _Float16* woxl = (_Float16*)(ws + OFF_WOXL);
  _Float16* woah = (_Float16*)(ws + OFF_WOAH);
  _Float16* woal = (_Float16*)(ws + OFF_WOAL);

  float* out   = (float*)d_out;
  float* g1out = out + (size_t)NN * FOUTN;

  prep_kernel<<<148, 256, 0, stream>>>(Wf1, Wf2, Wout,
                                       wf1h, wf1l, wf2h, wf2l,
                                       woxh, woxl, woah, woal);

  const int nblk = (NTILES + WPB - 1) / WPB;
  main_kernel<<<nblk, 32 * WPB, 0, stream>>>(x, bf1, bf2, tri_t, gauss_c, line_W, line_b,
                                             bout, fv,
                                             wf1h, wf1l, wf2h, wf2l,
                                             woxh, woxl, woah, woal, out);

  g1_kernel<<<BB, 32, 0, stream>>>(fv, edge_index, random_edges,
                                   tri_t1, gauss_c1, line_W1, line_b1, g1out);
}
